// QRNNLayer_45930380263758
// MI455X (gfx1250) — compile-verified, matrix operand order checked by reading, not run on hardware
//
#include <hip/hip_runtime.h>
#include <hip/hip_bf16.h>
#include <cstdint>

#define S_LEN  1024
#define BATCH  16
#define DDIM   1024
#define HDIM   1024
#define M_ROWS (S_LEN * BATCH)   // 16384
#define N_COLS (3 * HDIM)        // 3072
#define NCH    (BATCH * HDIM)    // 16384 channels
#define NCHUNK 8
#define CHUNK_LEN (S_LEN / NCHUNK)  // 128

typedef __bf16 bf16;
typedef __attribute__((ext_vector_type(8)))  __bf16 bh8;
typedef __attribute__((ext_vector_type(16))) __bf16 bh16;
typedef __attribute__((ext_vector_type(8)))  float  v8f;

__device__ __forceinline__ float fast_sigmoid(float x) {
    return 1.0f / (1.0f + __expf(-x));
}
__device__ __forceinline__ float fast_tanh(float x) {
    return 1.0f - 2.0f / (__expf(2.0f * x) + 1.0f);
}

// ---------------- pass 1: fp32 -> bf16 (round to nearest even) ----------------
__global__ void cvt_bf16_kernel(const float* __restrict__ in,
                                unsigned short* __restrict__ out, int n) {
    int stride = gridDim.x * blockDim.x;
    for (int i = blockIdx.x * blockDim.x + threadIdx.x; i < n; i += stride) {
        unsigned int u = __float_as_uint(in[i]);
        unsigned int r = u + 0x7FFFu + ((u >> 16) & 1u);
        out[i] = (unsigned short)(r >> 16);
    }
}

// ---------------- pass 2: bf16 WMMA GEMM + bias + gate activations ------------
union UB { bh16 v; bh8 h[2]; };
struct FragSet { UB a, b0, b1, b2, b3; };

__device__ __forceinline__ void load_set(FragSet& s,
    const bf16* __restrict__ aRow,
    const bf16* __restrict__ w0, const bf16* __restrict__ w1,
    const bf16* __restrict__ w2, const bf16* __restrict__ w3,
    int k0, int klo, int khi) {
    s.a.h[0]  = *reinterpret_cast<const bh8*>(aRow + k0 + klo);
    s.a.h[1]  = *reinterpret_cast<const bh8*>(aRow + k0 + khi);
    s.b0.h[0] = *reinterpret_cast<const bh8*>(w0 + k0 + klo);
    s.b0.h[1] = *reinterpret_cast<const bh8*>(w0 + k0 + khi);
    s.b1.h[0] = *reinterpret_cast<const bh8*>(w1 + k0 + klo);
    s.b1.h[1] = *reinterpret_cast<const bh8*>(w1 + k0 + khi);
    s.b2.h[0] = *reinterpret_cast<const bh8*>(w2 + k0 + klo);
    s.b2.h[1] = *reinterpret_cast<const bh8*>(w2 + k0 + khi);
    s.b3.h[0] = *reinterpret_cast<const bh8*>(w3 + k0 + klo);
    s.b3.h[1] = *reinterpret_cast<const bh8*>(w3 + k0 + khi);
}

__device__ __forceinline__ void mma_set(const FragSet& s,
    v8f& a0, v8f& a1, v8f& a2, v8f& a3) {
    a0 = __builtin_amdgcn_wmma_f32_16x16x32_bf16(false, s.a.v, false, s.b0.v,
                                                 (short)0, a0, false, false);
    a1 = __builtin_amdgcn_wmma_f32_16x16x32_bf16(false, s.a.v, false, s.b1.v,
                                                 (short)0, a1, false, false);
    a2 = __builtin_amdgcn_wmma_f32_16x16x32_bf16(false, s.a.v, false, s.b2.v,
                                                 (short)0, a2, false, false);
    a3 = __builtin_amdgcn_wmma_f32_16x16x32_bf16(false, s.a.v, false, s.b3.v,
                                                 (short)0, a3, false, false);
}

__global__ void __launch_bounds__(256)
qrnn_gemm_kernel(const bf16* __restrict__ Xb,    // [M_ROWS, DDIM]
                 const bf16* __restrict__ Wb,    // [N_COLS, DDIM]
                 const float* __restrict__ bias, // [N_COLS]
                 float* __restrict__ G)          // [3][M_ROWS*HDIM]
{
    const int lane = threadIdx.x & 31;
    const int wave = threadIdx.x >> 5;
    const int l16  = lane & 15;
    const int half = lane >> 4;

    const int mTile = blockIdx.y;                    // 0..1023 (16 rows each)
    const int nBase = blockIdx.x * 512 + wave * 64;  // multiple of 64; 64 | 1024

    const bf16* aRow = Xb + (size_t)(mTile * 16 + l16) * DDIM;
    const bf16* w0   = Wb + (size_t)(nBase +  0 + l16) * DDIM;
    const bf16* w1   = Wb + (size_t)(nBase + 16 + l16) * DDIM;
    const bf16* w2   = Wb + (size_t)(nBase + 32 + l16) * DDIM;
    const bf16* w3   = Wb + (size_t)(nBase + 48 + l16) * DDIM;

    v8f acc0 = {}, acc1 = {}, acc2 = {}, acc3 = {};

    // ISA 16-bit A/B fragment layout: lane half selects K groups
    const int klo = half * 8;
    const int khi = 16 + half * 8;

    // Software-pipelined, fully unrolled K loop. sched_barrier(0) fences pin
    // the prefetch loads BEFORE the current step's WMMAs so the scheduler
    // cannot collapse the double buffer; f0/f1 stay simultaneously live and
    // the wait in front of each WMMA group leaves the next step's 10 loads
    // in flight (s_wait_loadcnt 0xa instead of 0x0).
    FragSet f0, f1;
    load_set(f0, aRow, w0, w1, w2, w3, 0, klo, khi);
#pragma unroll
    for (int k0 = 0; k0 < DDIM; k0 += 64) {
        load_set(f1, aRow, w0, w1, w2, w3, k0 + 32, klo, khi);
        __builtin_amdgcn_sched_barrier(0);
        mma_set(f0, acc0, acc1, acc2, acc3);
        __builtin_amdgcn_sched_barrier(0);
        if (k0 + 64 < DDIM)
            load_set(f0, aRow, w0, w1, w2, w3, k0 + 64, klo, khi);
        __builtin_amdgcn_sched_barrier(0);
        mma_set(f1, acc0, acc1, acc2, acc3);
        __builtin_amdgcn_sched_barrier(0);
    }

    // Epilogue. gate is constant per wave (nBase % 64 == 0 and 64 | 1024).
    const int gate     = nBase / HDIM;   // 0 = Z, 1 = F, 2 = O
    const int hcolBase = nBase % HDIM;
    float* Gg = G + (size_t)gate * M_ROWS * HDIM;
    const int mBase = mTile * 16 + half * 8;

    const float bs0 = bias[nBase +  0 + l16];
    const float bs1 = bias[nBase + 16 + l16];
    const float bs2 = bias[nBase + 32 + l16];
    const float bs3 = bias[nBase + 48 + l16];

#pragma unroll
    for (int r = 0; r < 8; ++r) {
        // C/D layout: VGPR r holds row m = r (lanes 0-15) / m = 8+r (lanes 16-31)
        const size_t rowOff = (size_t)(mBase + r) * HDIM;
        float v0 = acc0[r] + bs0;
        float v1 = acc1[r] + bs1;
        float v2 = acc2[r] + bs2;
        float v3 = acc3[r] + bs3;
        if (gate == 0) {
            v0 = fast_tanh(v0); v1 = fast_tanh(v1);
            v2 = fast_tanh(v2); v3 = fast_tanh(v3);
        } else {
            v0 = fast_sigmoid(v0); v1 = fast_sigmoid(v1);
            v2 = fast_sigmoid(v2); v3 = fast_sigmoid(v3);
        }
        Gg[rowOff + hcolBase +  0 + l16] = v0;
        Gg[rowOff + hcolBase + 16 + l16] = v1;
        Gg[rowOff + hcolBase + 32 + l16] = v2;
        Gg[rowOff + hcolBase + 48 + l16] = v3;
    }
}

// ---------------- pass 3: chunked parallel scan over S ------------------------
__global__ void __launch_bounds__(256)
scan_local_kernel(const float* __restrict__ Z, const float* __restrict__ F,
                  float* __restrict__ Lend, float* __restrict__ Gend) {
    const int idx   = blockIdx.x * blockDim.x + threadIdx.x; // 0..NCHUNK*NCH-1
    const int chunk = idx >> 14;
    const int c     = idx & (NCH - 1);
    size_t off = (size_t)chunk * CHUNK_LEN * NCH + c;
    float h = 0.0f, g = 1.0f;
#pragma unroll 4
    for (int s = 0; s < CHUNK_LEN; ++s) {
        float z = Z[off], f = F[off];
        float om = 1.0f - f;
        h = f * z + om * h;
        g *= om;
        off += NCH;
    }
    Lend[idx] = h;
    Gend[idx] = g;
}

__global__ void __launch_bounds__(256)
scan_fix_kernel(const float* __restrict__ Lend, const float* __restrict__ Gend,
                float* __restrict__ Hin, float* __restrict__ out) {
    const int c = blockIdx.x * blockDim.x + threadIdx.x;  // 0..NCH-1
    float h = 0.0f;
#pragma unroll
    for (int ch = 0; ch < NCHUNK; ++ch) {
        Hin[ch * NCH + c] = h;
        h = Lend[ch * NCH + c] + Gend[ch * NCH + c] * h;
    }
    out[(size_t)S_LEN * NCH + c] = h;   // C[-1]
}

__global__ void __launch_bounds__(256)
scan_apply_kernel(const float* __restrict__ Z, const float* __restrict__ F,
                  const float* __restrict__ O, const float* __restrict__ Hin,
                  float* __restrict__ out) {
    const int idx   = blockIdx.x * blockDim.x + threadIdx.x;
    const int chunk = idx >> 14;
    const int c     = idx & (NCH - 1);
    float h = Hin[idx];
    size_t off = (size_t)chunk * CHUNK_LEN * NCH + c;
#pragma unroll 4
    for (int s = 0; s < CHUNK_LEN; ++s) {
        float z = Z[off], f = F[off], o = O[off];
        h = f * z + (1.0f - f) * h;
        out[off] = o * h;
        off += NCH;
    }
}

extern "C" void kernel_launch(void* const* d_in, const int* in_sizes, int n_in,
                              void* d_out, int out_size, void* d_ws, size_t ws_size,
                              hipStream_t stream) {
    const float* X = (const float*)d_in[0];  // [S, B, D]
    const float* W = (const float*)d_in[1];  // [3H, D]
    const float* b = (const float*)d_in[2];  // [3H]
    float* out = (float*)d_out;

    char* ws = (char*)d_ws;
    const size_t xbBytes = (size_t)M_ROWS * DDIM * sizeof(unsigned short); // 32 MB
    const size_t wbBytes = (size_t)N_COLS * DDIM * sizeof(unsigned short); //  6 MB
    const size_t gBytes  = 3ull * M_ROWS * HDIM * sizeof(float);           // 192 MB
    const size_t sBytes  = (size_t)NCHUNK * NCH * sizeof(float);           // 0.5 MB

    unsigned short* Xb = (unsigned short*)ws;
    unsigned short* Wb = (unsigned short*)(ws + xbBytes);
    float* G    = (float*)(ws + xbBytes + wbBytes);
    float* Lend = (float*)(ws + xbBytes + wbBytes + gBytes);
    float* Gend = (float*)(ws + xbBytes + wbBytes + gBytes + sBytes);
    float* Hin  = (float*)(ws + xbBytes + wbBytes + gBytes + 2 * sBytes);

    cvt_bf16_kernel<<<2048, 256, 0, stream>>>(X, Xb, M_ROWS * DDIM);
    cvt_bf16_kernel<<<512,  256, 0, stream>>>(W, Wb, N_COLS * DDIM);

    dim3 grid(N_COLS / 512, M_ROWS / 16);  // (6, 1024)
    qrnn_gemm_kernel<<<grid, 256, 0, stream>>>((const bf16*)Xb, (const bf16*)Wb, b, G);

    const float* Zp = G;
    const float* Fp = G + (size_t)M_ROWS * HDIM;
    const float* Op = G + 2 * (size_t)M_ROWS * HDIM;

    scan_local_kernel<<<(NCHUNK * NCH) / 256, 256, 0, stream>>>(Zp, Fp, Lend, Gend);
    scan_fix_kernel<<<NCH / 256, 256, 0, stream>>>(Lend, Gend, Hin, out);
    scan_apply_kernel<<<(NCHUNK * NCH) / 256, 256, 0, stream>>>(Zp, Fp, Op, Hin, out);
}
